// _Informer_90555090468948
// MI455X (gfx1250) — compile-verified, matrix operand order checked by reading, not run on hardware
//
#include <hip/hip_runtime.h>
#include <hip/hip_bf16.h>
#include <math.h>

// ---------------- model constants ----------------
#define BB   16
#define LEN  1024
#define LDE  512
#define DM   512
#define NH   8
#define DH   64
#define DFF  2048
#define PRED 256

typedef __attribute__((ext_vector_type(16))) __bf16 v16bf;
typedef __attribute__((ext_vector_type(8)))  float  v8f;

__device__ __forceinline__ __bf16 f2bf(float f) {
  union { float f; unsigned u; } v; v.f = f;
  unsigned r = (v.u + 0x7FFFu + ((v.u >> 16) & 1u)) >> 16;
  unsigned short h = (unsigned short)r;
  __bf16 o; __builtin_memcpy(&o, &h, 2);
  return o;
}

__device__ __forceinline__ unsigned pack2bf(float a, float b) {
  union { float f; unsigned u; } x, y; x.f = a; y.f = b;
  unsigned lo = (x.u + 0x7FFFu + ((x.u >> 16) & 1u)) >> 16;
  unsigned hi = (y.u + 0x7FFFu + ((y.u >> 16) & 1u)) >> 16;
  return (lo & 0xFFFFu) | (hi << 16);
}

__device__ __forceinline__ unsigned hash3(unsigned a, unsigned b, unsigned c) {
  unsigned x = a * 0x9E3779B1u ^ (b + 0x85EBCA77u) ^ (c * 0xC2B2AE3Du);
  x ^= x >> 16; x *= 0x7FEB352Du; x ^= x >> 15; x *= 0x846CA68Bu; x ^= x >> 16;
  return x;
}

// ---------------- weight repack: fp32 [K,N] -> bf16 [N,K] ----------------
__global__ void repack_w_kernel(const float* __restrict__ W, __bf16* __restrict__ out,
                                int K, int N)
{
  int idx = blockIdx.x * blockDim.x + threadIdx.x;
  if (idx >= K * N) return;
  int k = idx / N, n = idx - k * N;
  out[(size_t)n * K + k] = f2bf(W[idx]);
}

// conv weights (O,I,3) -> 3 bf16 matrices [N=O][K=I]
__global__ void repack_conv_kernel(const float* __restrict__ W, __bf16* __restrict__ out)
{
  int idx = blockIdx.x * blockDim.x + threadIdx.x;
  if (idx >= 3 * DM * DM) return;
  int t = idx / (DM * DM);
  int rem = idx - t * DM * DM;
  int o = rem / DM, i = rem - o * DM;
  out[idx] = f2bf(W[(size_t)o * DM * 3 + i * 3 + t]);   // out[t][o][i]
}

// ---------------- WMMA bf16 GEMM: C = act(accum*C + A@B + bias) ----------------
// A:  [M,K] fp32 (rows optionally circularly shifted within each length-Lrows batch)
// Wb: bf16 [N,K] (pre-transposed weights).
// 128x64 tile per block, BK=64 (2 wmma K-chunks), 8 waves;
// wave w: 16-row strip x 4 wmma col-tiles -> 8 WMMAs per barrier round.
// Requires M % 128 == 0, N % 64 == 0, K % 64 == 0 (true at every call site).
#define ASTR 72            // bf16 LDS row stride for 64-wide K panel (pad 8)
template<int ACT, int ACCUM, int HASBIAS>
__global__ __launch_bounds__(256)
void wmma_gemm_kernel(const float* __restrict__ A, const __bf16* __restrict__ Wb,
                      const float* __restrict__ bias, float* __restrict__ C,
                      int M, int N, int K, int Lrows, int shift)
{
  __shared__ unsigned AsU[128 * (ASTR / 2)];   // 128 rows x 64 bf16
  __shared__ unsigned BsU[64 * (ASTR / 2)];    // 64 n-rows x 64 bf16
  const __bf16* As = (const __bf16*)AsU;
  const __bf16* Bs = (const __bf16*)BsU;

  int tid  = threadIdx.x;
  int wv   = tid >> 5, lane = tid & 31;
  int mbase = blockIdx.y * 128;
  int nbase = blockIdx.x * 64;

  v8f acc0 = {}, acc1 = {}, acc2 = {}, acc3 = {};

  // A-load: each thread owns 32 consecutive fp32 of one row
  int ar = tid >> 1;             // 0..127
  int ac = (tid & 1) * 32;       // 0 or 32
  int am = mbase + ar;
  int bbatch = am / Lrows;
  int ll = am - bbatch * Lrows;
  ll += shift; if (ll < 0) ll += Lrows; else if (ll >= Lrows) ll -= Lrows;
  int asrc = bbatch * Lrows + ll;
  // B-load: each thread copies 16 bf16 (32B) of one n-row
  int br = tid >> 2;             // 0..63
  int bc = (tid & 3) * 16;       // bf16 offset 0,16,32,48

  int hi   = lane >> 4;
  int mrow = wv * 16 + (lane & 15);
  int nl   = lane & 15;

  for (int k0 = 0; k0 < K; k0 += 64) {
    const float* ap = A + (size_t)asrc * K + (k0 + ac);
    const unsigned* bp = (const unsigned*)(Wb + (size_t)(nbase + br) * K + (k0 + bc));
    if (k0 + 64 < K) {
      __builtin_prefetch((const void*)(ap + 64), 0, 3);
      __builtin_prefetch((const void*)(bp + 32), 0, 3);
    }
    // stage A (fp32 -> packed bf16 pairs -> 32-bit LDS stores)
    {
      unsigned* dst = &AsU[ar * (ASTR / 2) + (ac >> 1)];
      #pragma unroll
      for (int j = 0; j < 16; ++j) dst[j] = pack2bf(ap[2 * j], ap[2 * j + 1]);
    }
    // stage B (bf16 copy)
    {
      unsigned* dst = &BsU[br * (ASTR / 2) + (bc >> 1)];
      #pragma unroll
      for (int j = 0; j < 8; ++j) dst[j] = bp[j];
    }
    __syncthreads();

    #pragma unroll
    for (int c = 0; c < 2; ++c) {
      v16bf af, bf0, bf1, bf2, bf3;
      #pragma unroll
      for (int e = 0; e < 8; ++e) {
        int ka = c * 32 + hi * 8 + e, kb = c * 32 + 16 + hi * 8 + e;
        af[e]     = As[mrow * ASTR + ka];
        af[8 + e] = As[mrow * ASTR + kb];
        bf0[e]     = Bs[(nl)      * ASTR + ka];
        bf0[8 + e] = Bs[(nl)      * ASTR + kb];
        bf1[e]     = Bs[(nl + 16) * ASTR + ka];
        bf1[8 + e] = Bs[(nl + 16) * ASTR + kb];
        bf2[e]     = Bs[(nl + 32) * ASTR + ka];
        bf2[8 + e] = Bs[(nl + 32) * ASTR + kb];
        bf3[e]     = Bs[(nl + 48) * ASTR + ka];
        bf3[8 + e] = Bs[(nl + 48) * ASTR + kb];
      }
      acc0 = __builtin_amdgcn_wmma_f32_16x16x32_bf16(false, af, false, bf0, (short)0, acc0, false, false);
      acc1 = __builtin_amdgcn_wmma_f32_16x16x32_bf16(false, af, false, bf1, (short)0, acc1, false, false);
      acc2 = __builtin_amdgcn_wmma_f32_16x16x32_bf16(false, af, false, bf2, (short)0, acc2, false, false);
      acc3 = __builtin_amdgcn_wmma_f32_16x16x32_bf16(false, af, false, bf3, (short)0, acc3, false, false);
    }
    __syncthreads();
  }

  int nn = lane & 15;
  int mo = (lane >> 4) * 8;
  #pragma unroll
  for (int r = 0; r < 8; ++r) {
    int row = mbase + wv * 16 + mo + r;
    float vals[4] = { acc0[r], acc1[r], acc2[r], acc3[r] };
    #pragma unroll
    for (int nt = 0; nt < 4; ++nt) {
      int col = nbase + nt * 16 + nn;
      size_t i0 = (size_t)row * N + col;
      float v = vals[nt];
      if (ACCUM)   v += C[i0];
      if (HASBIAS) v += bias[col];
      if (ACT == 1) v = 0.5f * v * (1.0f + erff(v * 0.7071067811865475f));
      C[i0] = v;
    }
  }
}

// ---------------- data embedding: circular conv(CIN=7,k=3) + PE + time emb ----------------
__global__ void embed_kernel(const float* __restrict__ x, const float* __restrict__ xmark,
                             const float* __restrict__ convW, const float* __restrict__ timeW,
                             const float* __restrict__ timeb, float* __restrict__ out, int L)
{
  int idx = blockIdx.x * blockDim.x + threadIdx.x;
  int total = BB * L * DM;
  if (idx >= total) return;
  int d = idx & (DM - 1);
  int l = (idx / DM) % L;
  int b = idx / (DM * L);
  float acc = timeb[d];
  #pragma unroll
  for (int t = 0; t < 3; ++t) {
    int ls = l + t - 1; if (ls < 0) ls += L; else if (ls >= L) ls -= L;
    const float* xr = x + ((size_t)(b * L + ls)) * 7;
    #pragma unroll
    for (int c = 0; c < 7; ++c) acc += xr[c] * convW[d * 21 + c * 3 + t];
  }
  const float* mr = xmark + ((size_t)(b * L + l)) * 4;
  #pragma unroll
  for (int m = 0; m < 4; ++m) acc += mr[m] * timeW[m * DM + d];
  int i2 = d & ~1;
  float dv = expf(-(float)i2 * 0.0179889469f);   // ln(10000)/512
  float ang = (float)l * dv;
  acc += (d & 1) ? cosf(ang) : sinf(ang);
  out[idx] = acc;
}

// ---------------- ProbSparse: sampled sparsity measure M ----------------
__global__ __launch_bounds__(256)
void sample_m_kernel(const float* __restrict__ Q, const float* __restrict__ Km,
                     float* __restrict__ Mout, int Lq, int Lk, int U, int tag)
{
  int gw   = (blockIdx.x * blockDim.x + threadIdx.x) >> 5;
  int lane = threadIdx.x & 31;
  int total = BB * NH * Lq;
  if (gw >= total) return;
  int l  = gw % Lq;
  int bh = gw / Lq;
  int h = bh % NH, b = bh / NH;
  const float* qr = Q + ((size_t)(b * Lq + l)) * DM + h * DH;
  float q0 = qr[lane], q1 = qr[lane + 32];
  float mx = -3.0e38f, sm = 0.0f;
  for (int s = 0; s < U; ++s) {
    int ki = (int)(hash3((unsigned)l, (unsigned)s, (unsigned)tag) % (unsigned)Lk);
    const float* kr = Km + ((size_t)(b * Lk + ki)) * DM + h * DH;
    float p = q0 * kr[lane] + q1 * kr[lane + 32];
    #pragma unroll
    for (int off = 16; off > 0; off >>= 1) p += __shfl_xor(p, off, 32);
    mx = fmaxf(mx, p); sm += p;
  }
  if (lane == 0) Mout[gw] = mx - sm / (float)Lk;
}

// ---------------- top-k (k=U) per (b,h): iterative argmax ----------------
__global__ __launch_bounds__(256)
void topk_kernel(const float* __restrict__ Min, int* __restrict__ topk, int Lq, int U)
{
  int bh = blockIdx.x, tid = threadIdx.x;
  __shared__ float mv[1024];
  __shared__ float rv[256];
  __shared__ int   ri[256];
  for (int i = tid; i < Lq; i += 256) mv[i] = Min[(size_t)bh * Lq + i];
  __syncthreads();
  for (int it = 0; it < U; ++it) {
    float bv = -3.0e38f; int bi = 0;
    for (int i = tid; i < Lq; i += 256)
      if (mv[i] > bv) { bv = mv[i]; bi = i; }
    rv[tid] = bv; ri[tid] = bi;
    __syncthreads();
    for (int s = 128; s > 0; s >>= 1) {
      if (tid < s) {
        float ov = rv[tid + s]; int oi = ri[tid + s];
        if (ov > rv[tid] || (ov == rv[tid] && oi < ri[tid])) { rv[tid] = ov; ri[tid] = oi; }
      }
      __syncthreads();
    }
    if (tid == 0) { topk[bh * U + it] = ri[0]; mv[ri[0]] = -3.0e38f; }
    __syncthreads();
  }
}

// ---------------- mean of V over Lk per (b,h,d) ----------------
__global__ void vmean_kernel(const float* __restrict__ V, float* __restrict__ vm, int Lk)
{
  int bh = blockIdx.x, d = threadIdx.x;
  int h = bh % NH, b = bh / NH;
  float s = 0.0f;
  for (int k = 0; k < Lk; ++k) s += V[((size_t)(b * Lk + k)) * DM + h * DH + d];
  vm[bh * DH + d] = s / (float)Lk;
}

__global__ void ctx_broadcast_kernel(const float* __restrict__ vm, float* __restrict__ out, int Lq)
{
  int idx = blockIdx.x * blockDim.x + threadIdx.x;
  int total = BB * Lq * DM;
  if (idx >= total) return;
  int c = idx & (DM - 1);
  int b = idx / (DM * Lq);
  int h = c >> 6, d = c & 63;
  out[idx] = vm[(b * NH + h) * DH + d];
}

__global__ void ctx_cumsum_kernel(const float* __restrict__ V, float* __restrict__ out, int Lq)
{
  int t = blockIdx.x * blockDim.x + threadIdx.x;
  if (t >= BB * NH * DH) return;
  int d = t & 63, h = (t >> 6) & 7, b = t >> 9;
  float run = 0.0f;
  for (int l = 0; l < Lq; ++l) {
    size_t i = ((size_t)(b * Lq + l)) * DM + h * DH + d;
    run += V[i]; out[i] = run;
  }
}

// ---------------- top-u scores + softmax + A@V update ----------------
__global__ __launch_bounds__(256)
void scores_upd_kernel(const float* __restrict__ Q, const float* __restrict__ Km,
                       const float* __restrict__ V, const int* __restrict__ topk,
                       float* __restrict__ upd, int Lq, int Lk, int U, int masked)
{
  int blk = blockIdx.x;
  int u  = blk % U;
  int bh = blk / U;
  int h = bh % NH, b = bh / NH;
  int tid = threadIdx.x;
  int lq = topk[bh * U + u];
  __shared__ float qs[DH];
  __shared__ float sc[1024];
  __shared__ float red[256];
  if (tid < DH) qs[tid] = Q[((size_t)(b * Lq + lq)) * DM + h * DH + tid];
  __syncthreads();
  for (int k = tid; k < Lk; k += 256) {
    const float* kr = Km + ((size_t)(b * Lk + k)) * DM + h * DH;
    float s = 0.0f;
    #pragma unroll
    for (int d = 0; d < DH; ++d) s += qs[d] * kr[d];
    s *= 0.125f;
    if (masked && k > lq) s = -1.0e30f;
    sc[k] = s;
  }
  __syncthreads();
  float m = -3.0e38f;
  for (int k = tid; k < Lk; k += 256) m = fmaxf(m, sc[k]);
  red[tid] = m; __syncthreads();
  for (int s = 128; s > 0; s >>= 1) { if (tid < s) red[tid] = fmaxf(red[tid], red[tid + s]); __syncthreads(); }
  m = red[0]; __syncthreads();
  float ss = 0.0f;
  for (int k = tid; k < Lk; k += 256) { float e = expf(sc[k] - m); sc[k] = e; ss += e; }
  red[tid] = ss; __syncthreads();
  for (int s = 128; s > 0; s >>= 1) { if (tid < s) red[tid] += red[tid + s]; __syncthreads(); }
  float S = red[0]; __syncthreads();
  int d = tid & 63, part = tid >> 6;
  float acc = 0.0f;
  for (int k = part; k < Lk; k += 4)
    acc += sc[k] * V[((size_t)(b * Lk + k)) * DM + h * DH + d];
  red[part * 64 + d] = acc; __syncthreads();
  if (part == 0) {
    float tot = red[d] + red[64 + d] + red[128 + d] + red[192 + d];
    upd[((size_t)bh * U + u) * DH + d] = tot / S;
  }
}

__global__ void scatter_kernel(const int* __restrict__ topk, const float* __restrict__ upd,
                               float* __restrict__ out, int Lq, int U)
{
  int idx = blockIdx.x * blockDim.x + threadIdx.x;
  int total = BB * NH * U * DH;
  if (idx >= total) return;
  int d = idx & 63;
  int tmp = idx >> 6;
  int u = tmp % U;
  int bh = tmp / U;
  int h = bh % NH, b = bh / NH;
  int l = topk[bh * U + u];
  out[((size_t)(b * Lq + l)) * DM + h * DH + d] = upd[((size_t)bh * U + u) * DH + d];
}

// ---------------- y = LayerNorm(x + res) * g + b  (row = 512) ----------------
__global__ __launch_bounds__(256)
void add_ln_kernel(const float* __restrict__ x, const float* __restrict__ res,
                   const float* __restrict__ g, const float* __restrict__ bta,
                   float* __restrict__ y)
{
  int row = blockIdx.x, tid = threadIdx.x;
  size_t base = (size_t)row * DM;
  float a = x[base + tid]       + (res ? res[base + tid]       : 0.0f);
  float b = x[base + tid + 256] + (res ? res[base + tid + 256] : 0.0f);
  __shared__ float red[256];
  red[tid] = a + b; __syncthreads();
  for (int s = 128; s > 0; s >>= 1) { if (tid < s) red[tid] += red[tid + s]; __syncthreads(); }
  float mu = red[0] * (1.0f / DM); __syncthreads();
  float da = a - mu, db = b - mu;
  red[tid] = da * da + db * db; __syncthreads();
  for (int s = 128; s > 0; s >>= 1) { if (tid < s) red[tid] += red[tid + s]; __syncthreads(); }
  float inv = rsqrtf(red[0] * (1.0f / DM) + 1e-5f);
  y[base + tid]       = da * inv * g[tid]       + bta[tid];
  y[base + tid + 256] = db * inv * g[tid + 256] + bta[tid + 256];
}

// ---------------- conv_distil helpers ----------------
__global__ void bn_init_kernel(float* acc) {
  int i = blockIdx.x * blockDim.x + threadIdx.x;
  if (i < 2 * DM) acc[i] = 0.0f;
}

__global__ __launch_bounds__(256)
void bn_stats_kernel(const float* __restrict__ h, float* __restrict__ acc, int Mr)
{
  int nb = gridDim.x;
  int rp = (Mr + nb - 1) / nb;
  int r0 = blockIdx.x * rp;
  int r1 = r0 + rp; if (r1 > Mr) r1 = Mr;
  int c0 = threadIdx.x, c1 = threadIdx.x + 256;
  float s0 = 0, q0 = 0, s1 = 0, q1 = 0;
  for (int r = r0; r < r1; ++r) {
    float a = h[(size_t)r * DM + c0]; s0 += a; q0 += a * a;
    float b = h[(size_t)r * DM + c1]; s1 += b; q1 += b * b;
  }
  atomicAdd(&acc[c0], s0); atomicAdd(&acc[DM + c0], q0);
  atomicAdd(&acc[c1], s1); atomicAdd(&acc[DM + c1], q1);
}

__global__ void bn_elu_kernel(float* __restrict__ h, const float* __restrict__ acc,
                              const float* __restrict__ gamma, const float* __restrict__ beta, int Mr)
{
  int idx = blockIdx.x * blockDim.x + threadIdx.x;
  if (idx >= Mr * DM) return;
  int c = idx & (DM - 1);
  float mu  = acc[c] / (float)Mr;
  float var = acc[DM + c] / (float)Mr - mu * mu;
  float x = (h[idx] - mu) * rsqrtf(var + 1e-5f) * gamma[c] + beta[c];
  h[idx] = x > 0.0f ? x : (expf(x) - 1.0f);
}

__global__ void maxpool_kernel(const float* __restrict__ h, float* __restrict__ out, int Lin, int Lout)
{
  int idx = blockIdx.x * blockDim.x + threadIdx.x;
  int total = BB * Lout * DM;
  if (idx >= total) return;
  int c = idx & (DM - 1);
  int lo = (idx / DM) % Lout;
  int b = idx / (DM * Lout);
  float m = -3.0e38f;
  #pragma unroll
  for (int j = 0; j < 3; ++j) {
    int l = 2 * lo - 1 + j;
    if (l >= 0 && l < Lin) m = fmaxf(m, h[((size_t)(b * Lin + l)) * DM + c]);
  }
  out[idx] = m;
}

// ---------------- final projection (N=7, last PRED positions) ----------------
__global__ void proj_kernel(const float* __restrict__ dec, const float* __restrict__ pw,
                            const float* __restrict__ pb, float* __restrict__ out)
{
  int idx = blockIdx.x * blockDim.x + threadIdx.x;
  int total = BB * PRED * 7;
  if (idx >= total) return;
  int c = idx % 7;
  int lp = (idx / 7) % PRED;
  int b = idx / (7 * PRED);
  int l = LDE - PRED + lp;
  const float* row = dec + ((size_t)(b * LDE + l)) * DM;
  float acc = pb[c];
  for (int k = 0; k < DM; ++k) acc += row[k] * pw[k * 7 + c];
  out[idx] = acc;
}

// ================= parameter indices (pytree order: sorted dict keys) =================
enum {
  IN_XENC = 0, IN_XMARK_ENC, IN_XDEC, IN_XMARK_DEC,
  P_DEC_EMB_CONVW, P_DEC_EMB_TIMEW, P_DEC_EMB_TIMEB,                    // dec_emb
  P_DL_CR_WK, P_DL_CR_WO, P_DL_CR_WQ, P_DL_CR_WV,                       // dec_layers[0].cross
  P_DL_CR_BK, P_DL_CR_BO, P_DL_CR_BQ, P_DL_CR_BV,
  P_DL_FFN_W1, P_DL_FFN_W2, P_DL_FFN_B1, P_DL_FFN_B2,                   // .ffn
  P_DL_N1_B, P_DL_N1_G, P_DL_N2_B, P_DL_N2_G, P_DL_N3_B, P_DL_N3_G,     // .n1 .n2 .n3
  P_DL_SF_WK, P_DL_SF_WO, P_DL_SF_WQ, P_DL_SF_WV,                       // .self
  P_DL_SF_BK, P_DL_SF_BO, P_DL_SF_BQ, P_DL_SF_BV,
  P_DEC_NORM_B, P_DEC_NORM_G,
  P_CONV_W, P_CONV_B, P_CONV_BETA, P_CONV_GAMMA,                        // enc_convs[0]
  P_ENC_EMB_CONVW, P_ENC_EMB_TIMEW, P_ENC_EMB_TIMEB,
  P_EL0 = 42,   // enc layer: Wk,Wo,Wq,Wv,bk,bo,bq,bv,W1,W2,b1,b2,n1b,n1g,n2b,n2g
  P_EL1 = 58,
  P_ENC_NORM_B = 74, P_ENC_NORM_G, P_PROJ_W, P_PROJ_B
};

extern "C" void kernel_launch(void* const* d_in, const int* in_sizes, int n_in,
                              void* d_out, int out_size, void* d_ws, size_t ws_size,
                              hipStream_t stream)
{
  auto P = [&](int i) { return (const float*)d_in[i]; };
  float* ws = (float*)d_ws;

  const size_t ME = (size_t)BB * LEN * DM;           // 8M floats
  float* R0  = ws;                                   // activation stream
  float* R1  = ws + 1 * ME;                          // attn-ctx / ffn-out / conv-out
  float* R2  = ws + 2 * ME;                          // Q
  float* R3  = ws + 3 * ME;                          // K
  float* R4  = ws + 4 * ME;                          // V
  float* R5  = ws + 5 * ME;                          // attn projection out
  float* HF  = R2;                                   // FFN hidden aliases R2..R5 (4*ME)
  float* ENCB = ws + 6 * ME;                         // encoder output (kept for cross)
  __bf16* WBF  = (__bf16*)(ENCB + (size_t)BB * LDE * DM);  // repacked weight (<= 2048*512)
  __bf16* WBF3 = WBF + (size_t)DFF * DM;                   // 3 conv tap matrices
  float* MB  = (float*)(WBF3 + (size_t)3 * DM * DM);       // sparsity measure
  int*   TK  = (int*)(MB + (size_t)BB * NH * LEN);   // top-k indices (padded region)
  float* UPD = (float*)(TK + BB * NH * 64);          // top-u updates (U padded to 64)
  float* VM  = UPD + (size_t)BB * NH * 64 * DH;      // v-mean
  float* BNA = VM + BB * NH * DH;                    // BN accumulators (1024)

  const int U = 35;                                  // 5*ceil(ln L) for L in {512,1024}

  auto repack = [&](const float* W, int K, int N) {
    repack_w_kernel<<<(K * N + 255) / 256, 256, 0, stream>>>(W, WBF, K, N);
  };
  auto gemm = [&](const float* A, const __bf16* Wb, const float* bias, float* C,
                  int M, int N, int K, int act, int accum, int Lr, int sh) {
    dim3 g((unsigned)(N / 64), (unsigned)(M / 128));
    if (act == 1)
      wmma_gemm_kernel<1,0,1><<<g, 256, 0, stream>>>(A, Wb, bias, C, M, N, K, Lr, sh);
    else if (accum && bias)
      wmma_gemm_kernel<0,1,1><<<g, 256, 0, stream>>>(A, Wb, bias, C, M, N, K, Lr, sh);
    else if (accum)
      wmma_gemm_kernel<0,1,0><<<g, 256, 0, stream>>>(A, Wb, bias, C, M, N, K, Lr, sh);
    else if (bias)
      wmma_gemm_kernel<0,0,1><<<g, 256, 0, stream>>>(A, Wb, bias, C, M, N, K, Lr, sh);
    else
      wmma_gemm_kernel<0,0,0><<<g, 256, 0, stream>>>(A, Wb, bias, C, M, N, K, Lr, sh);
  };
  auto gemm_w = [&](const float* A, int widx, const float* bias, float* C,
                    int M, int N, int K, int act, int Lr) {
    repack(P(widx), K, N);
    gemm(A, WBF, bias, C, M, N, K, act, 0, Lr, 0);
  };

  auto attention = [&](const float* Xq, const float* Xkv, int Lq, int Lk,
                       int wk, int wo, int wq, int wv, int bk, int bo, int bq, int bv,
                       int masked, int tag) {
    int Mq = BB * Lq, Mk = BB * Lk;
    gemm_w(Xq,  wq, P(bq), R2, Mq, DM, DM, 0, Lq);
    gemm_w(Xkv, wk, P(bk), R3, Mk, DM, DM, 0, Lk);
    gemm_w(Xkv, wv, P(bv), R4, Mk, DM, DM, 0, Lk);
    int waves = BB * NH * Lq;
    sample_m_kernel<<<waves / 8, 256, 0, stream>>>(R2, R3, MB, Lq, Lk, U, tag);
    topk_kernel<<<BB * NH, 256, 0, stream>>>(MB, TK, Lq, U);
    if (masked) {
      ctx_cumsum_kernel<<<(BB * NH * DH + 255) / 256, 256, 0, stream>>>(R4, R1, Lq);
    } else {
      vmean_kernel<<<BB * NH, DH, 0, stream>>>(R4, VM, Lk);
      int tot = BB * Lq * DM;
      ctx_broadcast_kernel<<<(tot + 255) / 256, 256, 0, stream>>>(VM, R1, Lq);
    }
    scores_upd_kernel<<<BB * NH * U, 256, 0, stream>>>(R2, R3, R4, TK, UPD, Lq, Lk, U, masked);
    int ts = BB * NH * U * DH;
    scatter_kernel<<<(ts + 255) / 256, 256, 0, stream>>>(TK, UPD, R1, Lq, U);
    gemm_w(R1, wo, P(bo), R5, Mq, DM, DM, 0, Lq);
  };

  auto enc_layer = [&](int base, int L, int tag) {
    int Mr = BB * L;
    attention(R0, R0, L, L, base + 0, base + 1, base + 2, base + 3,
              base + 4, base + 5, base + 6, base + 7, 0, tag);
    add_ln_kernel<<<Mr, 256, 0, stream>>>(R0, R5, P(base + 13), P(base + 12), R0);
    gemm_w(R0, base + 8, P(base + 10), HF, Mr, DFF, DM, 1, L);   // GELU
    gemm_w(HF, base + 9, P(base + 11), R1, Mr, DM, DFF, 0, L);
    add_ln_kernel<<<Mr, 256, 0, stream>>>(R0, R1, P(base + 15), P(base + 14), R0);
  };

  // ---------------- encoder ----------------
  {
    int tot = BB * LEN * DM;
    embed_kernel<<<(tot + 255) / 256, 256, 0, stream>>>(
        (const float*)d_in[IN_XENC], (const float*)d_in[IN_XMARK_ENC],
        P(P_ENC_EMB_CONVW), P(P_ENC_EMB_TIMEW), P(P_ENC_EMB_TIMEB), R0, LEN);
  }
  enc_layer(P_EL0, LEN, 10);

  // conv_distil: circular conv(512->512,k=3) as 3 shifted GEMMs + BN + ELU + maxpool
  repack_conv_kernel<<<(3 * DM * DM + 255) / 256, 256, 0, stream>>>(P(P_CONV_W), WBF3);
  gemm(R0, WBF3,               nullptr,     R1, BB * LEN, DM, DM, 0, 0, LEN, -1);
  gemm(R0, WBF3 + DM * DM,     nullptr,     R1, BB * LEN, DM, DM, 0, 1, LEN,  0);
  gemm(R0, WBF3 + 2 * DM * DM, P(P_CONV_B), R1, BB * LEN, DM, DM, 0, 1, LEN,  1);
  bn_init_kernel<<<4, 256, 0, stream>>>(BNA);
  bn_stats_kernel<<<64, 256, 0, stream>>>(R1, BNA, BB * LEN);
  bn_elu_kernel<<<(BB * LEN * DM + 255) / 256, 256, 0, stream>>>(R1, BNA, P(P_CONV_GAMMA), P(P_CONV_BETA), BB * LEN);
  maxpool_kernel<<<(BB * LDE * DM + 255) / 256, 256, 0, stream>>>(R1, R0, LEN, LDE);

  enc_layer(P_EL1, LDE, 11);
  add_ln_kernel<<<BB * LDE, 256, 0, stream>>>(R0, nullptr, P(P_ENC_NORM_G), P(P_ENC_NORM_B), ENCB);

  // ---------------- decoder ----------------
  {
    int tot = BB * LDE * DM;
    embed_kernel<<<(tot + 255) / 256, 256, 0, stream>>>(
        (const float*)d_in[IN_XDEC], (const float*)d_in[IN_XMARK_DEC],
        P(P_DEC_EMB_CONVW), P(P_DEC_EMB_TIMEW), P(P_DEC_EMB_TIMEB), R0, LDE);
  }
  // masked self-attention
  attention(R0, R0, LDE, LDE, P_DL_SF_WK, P_DL_SF_WO, P_DL_SF_WQ, P_DL_SF_WV,
            P_DL_SF_BK, P_DL_SF_BO, P_DL_SF_BQ, P_DL_SF_BV, 1, 100);
  add_ln_kernel<<<BB * LDE, 256, 0, stream>>>(R0, R5, P(P_DL_N1_G), P(P_DL_N1_B), R0);
  // cross-attention against encoder output
  attention(R0, ENCB, LDE, LDE, P_DL_CR_WK, P_DL_CR_WO, P_DL_CR_WQ, P_DL_CR_WV,
            P_DL_CR_BK, P_DL_CR_BO, P_DL_CR_BQ, P_DL_CR_BV, 0, 101);
  add_ln_kernel<<<BB * LDE, 256, 0, stream>>>(R0, R5, P(P_DL_N2_G), P(P_DL_N2_B), R0);
  // FFN
  gemm_w(R0, P_DL_FFN_W1, P(P_DL_FFN_B1), HF, BB * LDE, DFF, DM, 1, LDE);
  gemm_w(HF, P_DL_FFN_W2, P(P_DL_FFN_B2), R1, BB * LDE, DM, DFF, 0, LDE);
  add_ln_kernel<<<BB * LDE, 256, 0, stream>>>(R0, R1, P(P_DL_N3_G), P(P_DL_N3_B), R0);
  // final norm + projection (last PRED positions)
  add_ln_kernel<<<BB * LDE, 256, 0, stream>>>(R0, nullptr, P(P_DEC_NORM_G), P(P_DEC_NORM_B), R1);
  proj_kernel<<<(BB * PRED * 7 + 255) / 256, 256, 0, stream>>>(R1, P(P_PROJ_W), P(P_PROJ_B), (float*)d_out);
}
